// DeepONetCfCDecoder_11751030522114
// MI455X (gfx1250) — matrix operand order repaired and checked by reading, not compiled, not run on hardware
//
#include <hip/hip_runtime.h>
#include <hip/hip_bf16.h>

// MI455X (gfx1250) DeepONet CfC decoder.
// Strategy:
//  * Hoist branch token/K/V projections from per-query (N=8192) to per-timestep
//    (T=512): 16x less matmul work (exact).
//  * rel-bias is provably constant (LN over a size-1 axis) -> softmax-invariant
//    -> eliminated exactly.
//  * All GEMMs on v_wmma_f32_16x16x32_f16 (f16 in / f32 accumulate).
//  * Weights pre-swizzled once into the CDNA5 B-fragment VGPR layout so every
//    fragment fetch is two aligned b128 LDS loads; staging is a pure b128 copy.

typedef __attribute__((ext_vector_type(16))) _Float16 v16h;
typedef __attribute__((ext_vector_type(8)))  _Float16 v8h;
typedef __attribute__((ext_vector_type(8)))  float    v8f;

#define T_STEPS 512

// ---------------------------------------------------------------------------
// Kernel 0: convert f32 weight [Kdim x N] into swizzled f16 fragment layout.
// dst halfword index t = (((nt*KKc + kk)*32 + lane)*16 + e)
// holds W[kk*32 + 8*(lane>>4) + (e&7) + 16*(e>>3)][nt*16 + (lane&15)]  (zero-padded K rows).
// ---------------------------------------------------------------------------
__global__ __launch_bounds__(256) void swizzle_w_kernel(const float* __restrict__ W,
                                                        _Float16* __restrict__ dst,
                                                        int Kdim, int KKc, int N) {
    int total = KKc * 32 * N;
    for (int t = blockIdx.x * blockDim.x + threadIdx.x; t < total;
         t += gridDim.x * blockDim.x) {
        int e    = t & 15;
        int lane = (t >> 4) & 31;
        int rest = t >> 9;
        int kk   = rest % KKc;
        int nt   = rest / KKc;
        int k = kk * 32 + ((lane >> 4) << 3) + (e & 7) + ((e >> 3) << 4);
        int n = nt * 16 + (lane & 15);
        float v = (k < Kdim) ? W[(size_t)k * N + n] : 0.f;
        dst[t] = (_Float16)v;
    }
}

// ---------------------------------------------------------------------------
// Helpers
// ---------------------------------------------------------------------------
__device__ __forceinline__ void stage_lds(_Float16* dst, const _Float16* src,
                                          int halves, int tid) {
    __syncthreads();                     // previous consumers of dst are done
    const int4* s = (const int4*)src;    // 16B = 8 halves
    int4* d = (int4*)dst;
    int n = halves >> 3;
    for (int i = tid; i < n; i += 256) d[i] = s[i];
    __syncthreads();
}

// One 16x16 output tile: A (row-major f16 in LDS, leading dim aldk halves,
// rows arow0..arow0+15) x swizzled weight chunk (nkk k-steps of 32).
__device__ __forceinline__ v8f wmma_tile16(const _Float16* A, int arow0, int aldk,
                                           const _Float16* Wc, int ntl, int nkk,
                                           int lane) {
    v8f acc = {};
    const int m = lane & 15, g = lane >> 4;
    const _Float16* ap = A + (size_t)(arow0 + m) * aldk + g * 8;
    const _Float16* bp = Wc + ((size_t)(ntl * nkk) * 32 + lane) * 16;
    for (int kk = 0; kk < nkk; ++kk) {
        union { v16h v; v8h h[2]; } ua, ub;
        ua.h[0] = *(const v8h*)(ap);        // K = kk*32 + g*8 + [0..7]
        ua.h[1] = *(const v8h*)(ap + 16);   // K = kk*32 + 16 + g*8 + [0..7]
        ub.h[0] = *(const v8h*)(bp);
        ub.h[1] = *(const v8h*)(bp + 8);
        acc = __builtin_amdgcn_wmma_f32_16x16x32_f16(false, ua.v, false, ub.v,
                                                     (short)0, acc, false, false);
        ap += 32;
        bp += 512;                          // next fragment of this (ntl) column
    }
    return acc;
}

// ---------------------------------------------------------------------------
// Kernel 1: per-timestep precompute.  BT = h[t] @ Wtok + b ;
// K[t] = BT @ Wk + b ; V[t] = BT @ Wv + b.   (M=64, N=256, K=256)
// ---------------------------------------------------------------------------
__global__ __launch_bounds__(256) void kv_kernel(
    const float* __restrict__ h_states,
    const _Float16* __restrict__ Wtok, const float* __restrict__ btok_b,
    const _Float16* __restrict__ Wk,   const float* __restrict__ bk_b,
    const _Float16* __restrict__ Wv,   const float* __restrict__ bv_b,
    float* __restrict__ Kout, float* __restrict__ Vout) {
    __shared__ __align__(16) _Float16 s_w[32768];     // 64KB weight chunk
    __shared__ __align__(16) _Float16 s_a[64 * 256];  // h[t] as f16
    __shared__ __align__(16) _Float16 s_bt[64 * 256]; // branch tokens f16

    const int t = blockIdx.x, tid = threadIdx.x;
    const int lane = tid & 31, wave = tid >> 5;

    // h_states[t] -> LDS f16 (row-major [token][d])
    const float4* src = (const float4*)(h_states + (size_t)t * 64 * 256);
    for (int i = tid; i < 64 * 256 / 4; i += 256) {
        float4 v = src[i];
        _Float16* d = s_a + i * 4;
        d[0] = (_Float16)v.x; d[1] = (_Float16)v.y;
        d[2] = (_Float16)v.z; d[3] = (_Float16)v.w;
    }

    // Stage 1: BT = h @ Wtok + btok_b   -> s_bt (f16)
    for (int ch = 0; ch < 2; ++ch) {
        stage_lds(s_w, Wtok + (size_t)ch * 8 * 8 * 512, 8 * 8 * 512, tid);
        for (int tt = wave; tt < 32; tt += 8) {
            int mt = tt >> 3, ntl = tt & 7;
            v8f acc = wmma_tile16(s_a, mt * 16, 256, s_w, ntl, 8, lane);
            int n = lane & 15, g = lane >> 4;
            int col = ch * 128 + ntl * 16 + n;
            float bb = btok_b[col];
            for (int r = 0; r < 8; ++r) {
                int row = mt * 16 + r + 8 * g;
                s_bt[row * 256 + col] = (_Float16)(acc[r] + bb);
            }
        }
    }

    // Stages 2/3: K = BT @ Wk + bk_b ; V = BT @ Wv + bv_b  -> global f32
    for (int which = 0; which < 2; ++which) {
        const _Float16* W = which ? Wv : Wk;
        const float* bias = which ? bv_b : bk_b;
        float* dst = (which ? Vout : Kout) + (size_t)t * 64 * 256;
        for (int ch = 0; ch < 2; ++ch) {
            stage_lds(s_w, W + (size_t)ch * 8 * 8 * 512, 8 * 8 * 512, tid);
            for (int tt = wave; tt < 32; tt += 8) {
                int mt = tt >> 3, ntl = tt & 7;
                v8f acc = wmma_tile16(s_bt, mt * 16, 256, s_w, ntl, 8, lane);
                int n = lane & 15, g = lane >> 4;
                int col = ch * 128 + ntl * 16 + n;
                float bb = bias[col];
                for (int r = 0; r < 8; ++r) {
                    int row = mt * 16 + r + 8 * g;
                    dst[row * 256 + col] = acc[r] + bb;
                }
            }
        }
    }
}

// ---------------------------------------------------------------------------
// Kernel 2: per-query decoder.  16 queries per block, 256 threads (8 waves).
// ---------------------------------------------------------------------------
__global__ __launch_bounds__(256) void decoder_kernel(
    const float* __restrict__ xy, const float* __restrict__ t_q,
    const int* __restrict__ c_arr, const float* __restrict__ sensor_time,
    const float* __restrict__ time_proj_w, const float* __restrict__ time_proj_b,
    const float* __restrict__ comp_emb,
    const _Float16* __restrict__ Wtr, const float* __restrict__ trunk_in_b,
    const float* __restrict__ bn_g, const float* __restrict__ bn_b,
    const _Float16* __restrict__ Wbq, const float* __restrict__ bq_b,
    const float* __restrict__ cln_g, const float* __restrict__ cln_b,
    const _Float16* __restrict__ Wc1, const float* __restrict__ cb1,
    const _Float16* __restrict__ Wc2, const float* __restrict__ cb2,
    const _Float16* __restrict__ Wto, const float* __restrict__ to_b,
    const _Float16* __restrict__ Wbp, const float* __restrict__ bp_b,
    const float* __restrict__ log_temp, const float* __restrict__ comp_scale,
    const float* __restrict__ comp_bias,
    const float* __restrict__ Kmat, const float* __restrict__ Vmat,
    float* __restrict__ out) {
    __shared__ __align__(16) _Float16 s_w[32768];      // 64KB weight chunk
    __shared__ __align__(16) _Float16 s_feat[16 * 96]; // trunk input (K padded 96)
    __shared__ __align__(16) _Float16 s_tf[16 * 256];  // trunk_feat f16
    __shared__ __align__(16) _Float16 s_xln[16 * 256]; // LN'd operand / ctx_fin
    __shared__ __align__(16) _Float16 s_mid[16 * 256]; // ctx MLP hidden
    __shared__ __align__(16) float    s_bufA[16 * 256]; // q  -> trunk_sel
    __shared__ __align__(16) float    s_bufB[16 * 256]; // ctx -> branch_sel
    __shared__ float s_att[16 * 64];
    __shared__ float s_red[256];
    __shared__ float s_red2[256];
    __shared__ float s_stat[32];
    __shared__ int   s_idx[16];
    __shared__ int   s_c[16];

    const int tid = threadIdx.x;
    const int lane = tid & 31, wave = tid >> 5;
    const int n0 = blockIdx.x * 16;

    // ---- Phase A: per-query scalar features -------------------------------
    if (tid < 16) {
        int qi = n0 + tid;
        float x = xy[2 * qi], y = xy[2 * qi + 1];
        float tq = t_q[qi];
        int cc = c_arr[qi];
        int lo = 0, hi = T_STEPS;
        while (lo < hi) {
            int mid = (lo + hi) >> 1;
            if (sensor_time[mid] <= tq) lo = mid + 1; else hi = mid;
        }
        int idx = lo - 1;
        if (idx < 0) idx = 0;
        if (idx > T_STEPS - 1) idx = T_STEPS - 1;
        float dt = tq - sensor_time[idx];
        if (dt < 0.f) dt = 0.f;
        _Float16* f = s_feat + tid * 96;
        const float TWO_PI = 6.28318530717958647692f;
        for (int h = 0; h < 8; ++h) {
            float fx = TWO_PI * x * (float)(h + 1);
            float fy = TWO_PI * y * (float)(h + 1);
            f[h]      = (_Float16)sinf(fx);
            f[8 + h]  = (_Float16)cosf(fx);
            f[16 + h] = (_Float16)sinf(fy);
            f[24 + h] = (_Float16)cosf(fy);
        }
        for (int j = 0; j < 32; ++j)
            f[32 + j] = (_Float16)(dt * time_proj_w[j] + time_proj_b[j]);
        for (int j = 0; j < 8; ++j)
            f[64 + j] = (_Float16)comp_emb[cc * 8 + j];
        for (int j = 72; j < 96; ++j) f[j] = (_Float16)0.f;
        s_idx[tid] = idx;
        s_c[tid] = cc;
    }

    // ---- Phase B: trunk_feat = silu(feat @ trunk_in_w + b)  (M16 K96 N256) --
    for (int ch = 0; ch < 2; ++ch) {
        stage_lds(s_w, Wtr + (size_t)ch * 8 * 3 * 512, 8 * 3 * 512, tid);
        v8f acc = wmma_tile16(s_feat, 0, 96, s_w, wave, 3, lane);
        int n = lane & 15, g = lane >> 4;
        int col = ch * 128 + wave * 16 + n;
        float bb = trunk_in_b[col];
        for (int r = 0; r < 8; ++r) {
            float v = acc[r] + bb;
            v = v / (1.f + expf(-v));
            s_tf[(r + 8 * g) * 256 + col] = (_Float16)v;
        }
    }
    __syncthreads();

    // ---- Phase C: LayerNorm(trunk_feat) -> s_xln --------------------------
    {
        int q = tid >> 4, j = tid & 15;
        float s = 0.f, s2 = 0.f;
        for (int e = 0; e < 16; ++e) {
            float v = (float)s_tf[q * 256 + j * 16 + e];
            s += v; s2 += v * v;
        }
        s_red[tid] = s; s_red2[tid] = s2;
        __syncthreads();
        if (j == 0) {
            float S = 0.f, S2 = 0.f;
            for (int e = 0; e < 16; ++e) { S += s_red[q * 16 + e]; S2 += s_red2[q * 16 + e]; }
            float mean = S * (1.f / 256.f);
            float var = S2 * (1.f / 256.f) - mean * mean;
            s_stat[2 * q] = mean;
            s_stat[2 * q + 1] = rsqrtf(var + 1e-5f);
        }
        __syncthreads();
        float mean = s_stat[2 * q], rstd = s_stat[2 * q + 1];
        for (int e = 0; e < 16; ++e) {
            int col = j * 16 + e;
            float v = (float)s_tf[q * 256 + col];
            s_xln[q * 256 + col] = (_Float16)(((v - mean) * rstd) * bn_g[col] + bn_b[col]);
        }
    }

    // ---- Phase D: q = LN(tf) @ bq_w + b  -> s_bufA (f32) ------------------
    for (int ch = 0; ch < 2; ++ch) {
        stage_lds(s_w, Wbq + (size_t)ch * 8 * 8 * 512, 8 * 8 * 512, tid);
        v8f acc = wmma_tile16(s_xln, 0, 256, s_w, wave, 8, lane);
        int n = lane & 15, g = lane >> 4;
        int col = ch * 128 + wave * 16 + n;
        float bb = bq_b[col];
        for (int r = 0; r < 8; ++r) s_bufA[(r + 8 * g) * 256 + col] = acc[r] + bb;
    }
    __syncthreads();

    // ---- Phase E: attention (f32 VALU; rel-bias constant -> dropped) ------
    for (int s = 0; s < 2; ++s) {
        int qq = wave * 2 + s;
        int tix = s_idx[qq];
        const float4* q4 = (const float4*)(s_bufA + qq * 256);
        const float4* Ka = (const float4*)(Kmat + ((size_t)tix * 64 + lane) * 256);
        const float4* Kb = (const float4*)(Kmat + ((size_t)tix * 64 + lane + 32) * 256);
        float sA = 0.f, sB = 0.f;
        for (int j = 0; j < 64; ++j) {
            float4 qv = q4[j];
            float4 a = Ka[j], b = Kb[j];
            sA += qv.x * a.x + qv.y * a.y + qv.z * a.z + qv.w * a.w;
            sB += qv.x * b.x + qv.y * b.y + qv.z * b.z + qv.w * b.w;
        }
        sA *= 0.0625f; sB *= 0.0625f;               // 1/sqrt(H)
        float m = fmaxf(sA, sB);
        for (int off = 16; off > 0; off >>= 1) m = fmaxf(m, __shfl_xor(m, off, 32));
        float eA = expf(sA - m), eB = expf(sB - m);
        float sum = eA + eB;
        for (int off = 16; off > 0; off >>= 1) sum += __shfl_xor(sum, off, 32);
        float inv = 1.f / sum;
        s_att[qq * 64 + lane] = eA * inv;
        s_att[qq * 64 + lane + 32] = eB * inv;
        // ctx = attn @ V : lane owns 8 dims, coalesced V reads
        float a0 = 0, a1 = 0, a2 = 0, a3 = 0, a4 = 0, a5 = 0, a6 = 0, a7 = 0;
        const float* V0 = Vmat + (size_t)tix * 64 * 256 + lane * 8;
        for (int k = 0; k < 64; ++k) {
            float w = s_att[qq * 64 + k];
            const float4* vr = (const float4*)(V0 + k * 256);
            float4 v0 = vr[0], v1 = vr[1];
            a0 += w * v0.x; a1 += w * v0.y; a2 += w * v0.z; a3 += w * v0.w;
            a4 += w * v1.x; a5 += w * v1.y; a6 += w * v1.z; a7 += w * v1.w;
        }
        float* cd = s_bufB + qq * 256 + lane * 8;
        cd[0] = a0; cd[1] = a1; cd[2] = a2; cd[3] = a3;
        cd[4] = a4; cd[5] = a5; cd[6] = a6; cd[7] = a7;
    }
    __syncthreads();

    // ---- Phase F: LayerNorm(ctx) -> s_xln ---------------------------------
    {
        int q = tid >> 4, j = tid & 15;
        float s = 0.f, s2 = 0.f;
        for (int e = 0; e < 16; ++e) {
            float v = s_bufB[q * 256 + j * 16 + e];
            s += v; s2 += v * v;
        }
        s_red[tid] = s; s_red2[tid] = s2;
        __syncthreads();
        if (j == 0) {
            float S = 0.f, S2 = 0.f;
            for (int e = 0; e < 16; ++e) { S += s_red[q * 16 + e]; S2 += s_red2[q * 16 + e]; }
            float mean = S * (1.f / 256.f);
            float var = S2 * (1.f / 256.f) - mean * mean;
            s_stat[2 * q] = mean;
            s_stat[2 * q + 1] = rsqrtf(var + 1e-5f);
        }
        __syncthreads();
        float mean = s_stat[2 * q], rstd = s_stat[2 * q + 1];
        for (int e = 0; e < 16; ++e) {
            int col = j * 16 + e;
            float v = s_bufB[q * 256 + col];
            s_xln[q * 256 + col] = (_Float16)(((v - mean) * rstd) * cln_g[col] + cln_b[col]);
        }
    }

    // ---- Phase G: ctx MLP (cw1 silu, cw2 + residual) -> ctx_fin in s_xln --
    for (int ch = 0; ch < 2; ++ch) {
        stage_lds(s_w, Wc1 + (size_t)ch * 8 * 8 * 512, 8 * 8 * 512, tid);
        v8f acc = wmma_tile16(s_xln, 0, 256, s_w, wave, 8, lane);
        int n = lane & 15, g = lane >> 4;
        int col = ch * 128 + wave * 16 + n;
        float bb = cb1[col];
        for (int r = 0; r < 8; ++r) {
            float v = acc[r] + bb;
            v = v / (1.f + expf(-v));
            s_mid[(r + 8 * g) * 256 + col] = (_Float16)v;
        }
    }
    for (int ch = 0; ch < 2; ++ch) {
        stage_lds(s_w, Wc2 + (size_t)ch * 8 * 8 * 512, 8 * 8 * 512, tid);
        v8f acc = wmma_tile16(s_mid, 0, 256, s_w, wave, 8, lane);
        int n = lane & 15, g = lane >> 4;
        int col = ch * 128 + wave * 16 + n;
        float bb = cb2[col];
        for (int r = 0; r < 8; ++r) {
            int row = r + 8 * g;
            s_xln[row * 256 + col] = (_Float16)(acc[r] + bb + s_bufB[row * 256 + col]);
        }
    }

    // ---- Phase H: component-selected basis projections --------------------
    for (int cc = 0; cc < 3; ++cc) {
        for (int ch = 0; ch < 2; ++ch) {
            int nt0 = cc * 16 + ch * 8;
            // trunk_sel chunk
            stage_lds(s_w, Wto + (size_t)nt0 * 8 * 512, 8 * 8 * 512, tid);
            v8f acc = wmma_tile16(s_tf, 0, 256, s_w, wave, 8, lane);
            int n = lane & 15, g = lane >> 4;
            int col = ch * 128 + wave * 16 + n;
            float bb = to_b[cc * 256 + col];
            for (int r = 0; r < 8; ++r) {
                int row = r + 8 * g;
                if (s_c[row] == cc) s_bufA[row * 256 + col] = acc[r] + bb;
            }
            // branch_sel chunk
            stage_lds(s_w, Wbp + (size_t)nt0 * 8 * 512, 8 * 8 * 512, tid);
            acc = wmma_tile16(s_xln, 0, 256, s_w, wave, 8, lane);
            bb = bp_b[cc * 256 + col];
            for (int r = 0; r < 8; ++r) {
                int row = r + 8 * g;
                if (s_c[row] == cc) s_bufB[row * 256 + col] = acc[r] + bb;
            }
        }
    }
    __syncthreads();

    // ---- Phase I: out = <trunk_sel, branch_sel> * temp * scale + bias -----
    {
        int q = tid >> 4, j = tid & 15;
        float s = 0.f;
        for (int e = 0; e < 16; ++e) {
            int col = j * 16 + e;
            s += s_bufA[q * 256 + col] * s_bufB[q * 256 + col];
        }
        s_red[tid] = s;
        __syncthreads();
        if (j == 0) {
            float S = 0.f;
            for (int e = 0; e < 16; ++e) S += s_red[q * 16 + e];
            int cc = s_c[q];
            out[n0 + q] = S * expf(log_temp[0]) * comp_scale[cc] + comp_bias[cc];
        }
    }
}

// ---------------------------------------------------------------------------
extern "C" void kernel_launch(void* const* d_in, const int* in_sizes, int n_in,
                              void* d_out, int out_size, void* d_ws, size_t ws_size,
                              hipStream_t stream) {
    (void)n_in; (void)out_size; (void)ws_size;
    const float* xy          = (const float*)d_in[0];
    const float* t_q         = (const float*)d_in[1];
    const int*   c_arr       = (const int*)d_in[2];
    const float* h_states    = (const float*)d_in[3];
    const float* sensor_time = (const float*)d_in[4];
    // d_in[5] sensor_pos: unused (rel-bias is provably constant)
    const float* time_proj_w = (const float*)d_in[6];
    const float* time_proj_b = (const float*)d_in[7];
    const float* comp_emb    = (const float*)d_in[8];
    const float* trunk_in_w  = (const float*)d_in[9];
    const float* trunk_in_b  = (const float*)d_in[10];
    const float* bn_g        = (const float*)d_in[11];
    const float* bn_b        = (const float*)d_in[12];
    const float* btok_w      = (const float*)d_in[13];
    const float* btok_b      = (const float*)d_in[14];
    const float* bq_w        = (const float*)d_in[15];
    const float* bq_b        = (const float*)d_in[16];
    const float* bk_w        = (const float*)d_in[17];
    const float* bk_b        = (const float*)d_in[18];
    const float* bv_w        = (const float*)d_in[19];
    const float* bv_b        = (const float*)d_in[20];
    // d_in[21..26] rln_g, rln_b, rw1, rb1, rw2, rb2: unused (softmax-invariant)
    const float* cln_g       = (const float*)d_in[27];
    const float* cln_b       = (const float*)d_in[28];
    const float* cw1         = (const float*)d_in[29];
    const float* cb1         = (const float*)d_in[30];
    const float* cw2         = (const float*)d_in[31];
    const float* cb2         = (const float*)d_in[32];
    const float* to_w        = (const float*)d_in[33];
    const float* to_b        = (const float*)d_in[34];
    const float* bp_w        = (const float*)d_in[35];
    const float* bp_b        = (const float*)d_in[36];
    const float* log_temp    = (const float*)d_in[37];
    const float* comp_scale  = (const float*)d_in[38];
    const float* comp_bias   = (const float*)d_in[39];

    char* ws = (char*)d_ws;
    float* Kmat = (float*)ws;                                   // 32 MB
    float* Vmat = (float*)(ws + (size_t)T_STEPS * 64 * 256 * 4); // 32 MB
    size_t o = (size_t)T_STEPS * 64 * 256 * 4 * 2;
    _Float16* Wtok = (_Float16*)(ws + o); o += (size_t)256 * 256 * 2;
    _Float16* Wk   = (_Float16*)(ws + o); o += (size_t)256 * 256 * 2;
    _Float16* Wv   = (_Float16*)(ws + o); o += (size_t)256 * 256 * 2;
    _Float16* Wtr  = (_Float16*)(ws + o); o += (size_t)96 * 256 * 2;
    _Float16* Wbq  = (_Float16*)(ws + o); o += (size_t)256 * 256 * 2;
    _Float16* Wc1  = (_Float16*)(ws + o); o += (size_t)256 * 256 * 2;
    _Float16* Wc2  = (_Float16*)(ws + o); o += (size_t)256 * 256 * 2;
    _Float16* Wto  = (_Float16*)(ws + o); o += (size_t)256 * 768 * 2;
    _Float16* Wbp  = (_Float16*)(ws + o); o += (size_t)256 * 768 * 2;

    auto sw = [&](const float* W, _Float16* dst, int Kdim, int KKc, int N) {
        int total = KKc * 32 * N;
        swizzle_w_kernel<<<(total + 255) / 256, 256, 0, stream>>>(W, dst, Kdim, KKc, N);
    };
    sw(btok_w,     Wtok, 256, 8, 256);
    sw(bk_w,       Wk,   256, 8, 256);
    sw(bv_w,       Wv,   256, 8, 256);
    sw(trunk_in_w, Wtr,  72,  3, 256);
    sw(bq_w,       Wbq,  256, 8, 256);
    sw(cw1,        Wc1,  256, 8, 256);
    sw(cw2,        Wc2,  256, 8, 256);
    sw(to_w,       Wto,  256, 8, 768);
    sw(bp_w,       Wbp,  256, 8, 768);

    kv_kernel<<<T_STEPS, 256, 0, stream>>>(h_states, Wtok, btok_b, Wk, bk_b,
                                           Wv, bv_b, Kmat, Vmat);

    int nq = in_sizes[1];  // N = 8192
    decoder_kernel<<<nq / 16, 256, 0, stream>>>(
        xy, t_q, c_arr, sensor_time, time_proj_w, time_proj_b, comp_emb,
        Wtr, trunk_in_b, bn_g, bn_b, Wbq, bq_b, cln_g, cln_b,
        Wc1, cb1, Wc2, cb2, Wto, to_b, Wbp, bp_b,
        log_temp, comp_scale, comp_bias, Kmat, Vmat, (float*)d_out);
}
